// RNNDecoder_84052509983543
// MI455X (gfx1250) — matrix operand order repaired and checked by reading, not compiled, not run on hardware
//
#include <hip/hip_runtime.h>
#include <math.h>

#define B_  8
#define T_  13
#define N_  128
#define D_  4
#define H_  64
#define MH_ 128

typedef __attribute__((ext_vector_type(16))) _Float16 v16h;
typedef __attribute__((ext_vector_type(8)))  _Float16 h8;
typedef __attribute__((ext_vector_type(2)))  _Float16 h2;
typedef __attribute__((ext_vector_type(8)))  float    v8f;
typedef __attribute__((ext_vector_type(4)))  float    f4;

#if __has_builtin(__builtin_amdgcn_tanhf)
#define FAST_TANH(x) __builtin_amdgcn_tanhf(x)
#elif __has_builtin(__builtin_amdgcn_tanh_f32)
#define FAST_TANH(x) __builtin_amdgcn_tanh_f32(x)
#else
#define FAST_TANH(x) tanhf(x)
#endif

__device__ __forceinline__ float sigmoidf_(float x) { return 1.0f / (1.0f + __expf(-x)); }

// ---- WMMA fragment loaders (wave32 layouts per CDNA5 ISA 7.12.2) ----
// A (16x32 f16): lane L holds row M=L%16; its 16 halves are two contiguous
// K-runs of 8: K = kBase + 8*half + {0..7} and + {16..23}.  -> 2x b128 loads.
__device__ __forceinline__ v16h load_a_f16v(const _Float16* src, int ld, int rBase, int kBase, int lane) {
  int row  = rBase + (lane & 15);
  int half = (lane >> 4) & 1;
  const _Float16* p = src + row * ld + kBase + 8 * half;
  h8 lo = *(const h8*)p;
  h8 hi = *(const h8*)(p + 16);
  v16h a;
#pragma unroll
  for (int e = 0; e < 8; ++e) { a[e] = lo[e]; a[8 + e] = hi[e]; }
  return a;
}

// Same A layout but f32 source, converted on load (two pairs of b128 loads).
__device__ __forceinline__ v16h load_a_f32v(const float* src, int ld, int rBase, int kBase, int lane) {
  int row  = rBase + (lane & 15);
  int half = (lane >> 4) & 1;
  const float* p = src + row * ld + kBase + 8 * half;
  f4 x0 = *(const f4*)p;
  f4 x1 = *(const f4*)(p + 4);
  f4 y0 = *(const f4*)(p + 16);
  f4 y1 = *(const f4*)(p + 20);
  v16h a;
#pragma unroll
  for (int e = 0; e < 4; ++e) {
    a[e]      = (_Float16)x0[e];
    a[4 + e]  = (_Float16)x1[e];
    a[8 + e]  = (_Float16)y0[e];
    a[12 + e] = (_Float16)y1[e];
  }
  return a;
}

// B (32x16 f16): lane L holds column N=L%16; same K order as A: K = kBase + 8*half + {0..7} and + {16..23}.
// Source is TRANSPOSED (N-major: srcT[n*ldk + k]) so the run is contiguous.
__device__ __forceinline__ v16h load_b_tr(const _Float16* srcT, int ldk, int kBase, int nBase, int lane) {
  int col  = nBase + (lane & 15);
  int kOff = kBase + ((lane >> 4) & 1) * 8;
  const _Float16* p = srcT + col * ldk + kOff;
  h8 lo = *(const h8*)p;
  h8 hi = *(const h8*)(p + 16);
  v16h b;
#pragma unroll
  for (int e = 0; e < 8; ++e) { b[e] = lo[e]; b[8 + e] = hi[e]; }
  return b;
}

// ---- kernels ----
__global__ void k_init(float* h, float* sqacc) {
  int idx = blockIdx.x * blockDim.x + threadIdx.x;
  if (idx < B_ * N_ * H_) h[idx] = 0.0f;
  if (idx == 0) *sqacc = 0.0f;
}

// Wm1 (128x128 row-major, K x N) -> w1t[(uv*128 + n)*64 + k] = Wm1[(uv*64+k)*128 + n]
__global__ void k_prep(const float* Wm1, _Float16* w1t) {
  int idx = blockIdx.x * blockDim.x + threadIdx.x;
  if (idx < 2 * H_ * MH_) {
    int uv  = idx >> 13;
    int rem = idx & 8191;
    int n   = rem >> 6;
    int k   = rem & 63;
    w1t[idx] = (_Float16)Wm1[(uv * H_ + k) * MH_ + n];
  }
}

// Ap[b,i,j] = sigmoid(0.5*(A+A^T)) with zero diag, row-normalized over j.
__global__ void k_proj(const float* A, float* Ap) {
  int i = blockIdx.x, b = blockIdx.y, j = threadIdx.x;
  __shared__ float red[N_];
  const float* Ab = A + b * N_ * N_;
  float v = 0.0f;
  if (i != j) {
    float s = 0.5f * (Ab[i * N_ + j] + Ab[j * N_ + i]);
    v = 1.0f / (1.0f + __expf(-s));
  }
  red[j] = v;
  __syncthreads();
  for (int s = 64; s > 0; s >>= 1) {
    if (j < s) red[j] += red[j + s];
    __syncthreads();
  }
  Ap[b * N_ * N_ + i * N_ + j] = v / (red[0] + 1e-6f);
}

// U = h @ Wm1[0:64,:], V = h @ Wm1[64:128,:] ; one 16x16 C tile per wave, K=64.
// Outputs stored f16 (consumed as f16 by k_msg anyway).
__global__ void __launch_bounds__(256) k_uv(const float* h, const _Float16* w1t,
                                            _Float16* U, _Float16* V) {
  int lane = threadIdx.x & 31;
  int wave = threadIdx.x >> 5;
  int job  = blockIdx.x * 8 + wave;   // 0..1023
  int uv   = job & 1;
  int nt   = (job >> 1) & 7;
  int mt   = job >> 4;                // 0..63
  int mBase = mt * 16, nBase = nt * 16;
  const _Float16* wsel = w1t + uv * MH_ * H_;
  v8f acc = {};
#pragma unroll
  for (int kt = 0; kt < 2; ++kt) {
    v16h a  = load_a_f32v(h, H_, mBase, kt * 32, lane);
    v16h bf = load_b_tr(wsel, H_, kt * 32, nBase, lane);
    acc = __builtin_amdgcn_wmma_f32_16x16x32_f16(false, a, false, bf, (short)0, acc, false, false);
  }
  _Float16* dst = uv ? V : U;
  int col   = nBase + (lane & 15);
  int rhalf = ((lane >> 4) & 1) * 8;
#pragma unroll
  for (int r = 0; r < 8; ++r) dst[(mBase + rhalf + r) * MH_ + col] = (_Float16)acc[r];
}

// Per (b,j): Z = tanh(U_i + V_j + bm1) [128x128 f16 in LDS];
// M = tanh(Z @ Wm2 + bm2); agg[b,j] = sum_i Ap[b,i,j] * M[i].
__global__ void __launch_bounds__(256) k_msg(const _Float16* U, const _Float16* V, const float* Ap,
                                             const float* bm1, const float* Wm2, const float* bm2,
                                             float* agg) {
  int j = blockIdx.x, b = blockIdx.y;
  int tid = threadIdx.x;
  __shared__ _Float16 zs[N_ * MH_];    // 32 KB
  __shared__ _Float16 w2t[H_ * MH_];   // 16 KB, transposed: w2t[n*128 + k]
  __shared__ float vjb[MH_];
  __shared__ float acol[N_];
  __shared__ float part[8 * H_];       // per-wave partial agg

  const _Float16* Ub = U + (size_t)b * N_ * MH_;
  __builtin_prefetch(Ub + tid * 32, 0, 0);   // global_prefetch of the U slab

  if (tid < MH_) vjb[tid]  = (float)V[(b * N_ + j) * MH_ + tid] + bm1[tid];
  if (tid < N_)  acol[tid] = Ap[b * N_ * N_ + tid * N_ + j];
  // transpose Wm2 (KxN row-major) into N-major LDS; contiguous LDS writes
  for (int idx = tid; idx < H_ * MH_; idx += 256) {
    int n = idx >> 7, k = idx & (MH_ - 1);
    w2t[idx] = (_Float16)Wm2[k * H_ + n];
  }
  __syncthreads();

  // Z build: packed-pair f16 reads, hardware tanh, packed f16 LDS writes
  const h2* Ub2 = (const h2*)Ub;
  h2* zs2 = (h2*)zs;
  for (int idx = tid; idx < (N_ * MH_) / 2; idx += 256) {
    int m2 = idx & (MH_ / 2 - 1);
    h2 u = Ub2[idx];
    h2 z;
    z[0] = (_Float16)FAST_TANH((float)u[0] + vjb[2 * m2]);
    z[1] = (_Float16)FAST_TANH((float)u[1] + vjb[2 * m2 + 1]);
    zs2[idx] = z;
  }
  __syncthreads();

  int lane = tid & 31, wave = tid >> 5;
  int iBase = wave * 16;
  int col16 = lane & 15, half = (lane >> 4) & 1;

  // A fragments depend only on kt -> load once, reuse across all 4 n-tiles.
  v16h afrag[4];
#pragma unroll
  for (int kt = 0; kt < 4; ++kt) afrag[kt] = load_a_f16v(zs, MH_, iBase, kt * 32, lane);

#pragma unroll
  for (int ntile = 0; ntile < 4; ++ntile) {
    v8f acc = {};
#pragma unroll
    for (int kt = 0; kt < 4; ++kt) {
      v16h bf = load_b_tr(w2t, MH_, kt * 32, ntile * 16, lane);
      acc = __builtin_amdgcn_wmma_f32_16x16x32_f16(false, afrag[kt], false, bf, (short)0, acc, false, false);
    }
    int col = ntile * 16 + col16;
    float bias = bm2[col];
    float s = 0.0f;
#pragma unroll
    for (int r = 0; r < 8; ++r) {
      int Mrow = iBase + half * 8 + r;
      s += FAST_TANH(acc[r] + bias) * acol[Mrow];
    }
    s += __shfl_xor(s, 16, 32);              // combine the two row-halves
    if (half == 0) part[wave * H_ + col] = s;
  }
  __syncthreads();
  if (tid < H_) {
    float sum = 0.0f;
#pragma unroll
    for (int w = 0; w < 8; ++w) sum += part[w * H_ + tid];
    agg[(b * N_ + j) * H_ + tid] = sum;
  }
}

// GRU gate update + output MLP + squared-error accumulation. One block per node.
__global__ void __launch_bounds__(64) k_gru(const float* X, float* h, const float* agg,
    const float* Wir, const float* bir, const float* Wii, const float* bii,
    const float* Win, const float* bin_,
    const float* Whr, const float* Whi, const float* Whh,
    const float* Wmr, const float* Wmi, const float* Wmn,
    const float* Wo1, const float* bo1, const float* Wo2, const float* bo2,
    const float* Wo3, const float* bo3,
    float* Xpred, float* sqacc, int t) {
  int n = blockIdx.x, b = blockIdx.y, tid = threadIdx.x;
  __shared__ float hs[H_], as_[H_], xs[D_], hn[H_], p1[H_], p2[H_], sq[D_];
  int node = b * N_ + n;
  hs[tid]  = h[node * H_ + tid];
  as_[tid] = agg[node * H_ + tid];
  if (tid < D_) xs[tid] = X[((b * T_ + t) * N_ + n) * D_ + tid];
  __syncthreads();

  float xr = 0, xi = 0, xn = 0;
#pragma unroll
  for (int k = 0; k < D_; ++k) {
    float xv = xs[k];
    xr += xv * Wir[k * H_ + tid];
    xi += xv * Wii[k * H_ + tid];
    xn += xv * Win[k * H_ + tid];
  }
  float hr = 0, hi = 0, hh = 0, ar = 0, ai = 0, an = 0;
  for (int k = 0; k < H_; ++k) {
    float hv = hs[k], av = as_[k];
    hr += hv * Whr[k * H_ + tid];
    hi += hv * Whi[k * H_ + tid];
    hh += hv * Whh[k * H_ + tid];
    ar += av * Wmr[k * H_ + tid];
    ai += av * Wmi[k * H_ + tid];
    an += av * Wmn[k * H_ + tid];
  }
  float r  = sigmoidf_(xr + bir[tid] + hr + ar);
  float ig = sigmoidf_(xi + bii[tid] + hi + ai);
  float nn = FAST_TANH(xn + bin_[tid] + r * hh + an);
  float hnew = (1.0f - ig) * nn + ig * hs[tid];
  h[node * H_ + tid] = hnew;
  hn[tid] = hnew;
  __syncthreads();

  float a1 = bo1[tid];
  for (int k = 0; k < H_; ++k) a1 += hn[k] * Wo1[k * H_ + tid];
  p1[tid] = fmaxf(a1, 0.0f);
  __syncthreads();
  float a2 = bo2[tid];
  for (int k = 0; k < H_; ++k) a2 += p1[k] * Wo2[k * H_ + tid];
  p2[tid] = fmaxf(a2, 0.0f);
  __syncthreads();
  if (tid < D_) {
    float pv = xs[tid] + bo3[tid];
    for (int k = 0; k < H_; ++k) pv += p2[k] * Wo3[k * D_ + tid];
    Xpred[((b * (T_ - 1) + t) * N_ + n) * D_ + tid] = pv;
    float tgt = X[((b * T_ + (t + 1)) * N_ + n) * D_ + tid];
    float d = tgt - pv;
    sq[tid] = d * d;
  }
  __syncthreads();
  if (tid == 0) atomicAdd(sqacc, sq[0] + sq[1] + sq[2] + sq[3]);
}

__global__ void k_final(const float* log_sigma, const float* sqacc, float* out_ll) {
  float sigma = __expf(log_sigma[0]);
  sigma = fminf(fmaxf(sigma, 1e-4f), 10.0f);
  float cst = 0.5f * (float)(B_ * N_ * D_) * logf(6.283185307179586f * sigma * sigma);
  float inv2s2 = 1.0f / (2.0f * sigma * sigma);
  out_ll[0] = -((float)(T_ - 1) * cst + sqacc[0] * inv2s2);
}

extern "C" void kernel_launch(void* const* d_in, const int* in_sizes, int n_in,
                              void* d_out, int out_size, void* d_ws, size_t ws_size,
                              hipStream_t stream) {
  const float* A    = (const float*)d_in[0];
  const float* X    = (const float*)d_in[1];
  const float* Wm1  = (const float*)d_in[2];
  const float* bm1  = (const float*)d_in[3];
  const float* Wm2  = (const float*)d_in[4];
  const float* bm2  = (const float*)d_in[5];
  const float* Wir  = (const float*)d_in[6];
  const float* bir  = (const float*)d_in[7];
  const float* Wii  = (const float*)d_in[8];
  const float* bii  = (const float*)d_in[9];
  const float* Win  = (const float*)d_in[10];
  const float* bin_ = (const float*)d_in[11];
  const float* Whr  = (const float*)d_in[12];
  const float* Whi  = (const float*)d_in[13];
  const float* Whh  = (const float*)d_in[14];
  const float* Wmr  = (const float*)d_in[15];
  const float* Wmi  = (const float*)d_in[16];
  const float* Wmn  = (const float*)d_in[17];
  const float* Wo1  = (const float*)d_in[18];
  const float* bo1  = (const float*)d_in[19];
  const float* Wo2  = (const float*)d_in[20];
  const float* bo2  = (const float*)d_in[21];
  const float* Wo3  = (const float*)d_in[22];
  const float* bo3  = (const float*)d_in[23];
  const float* log_sigma = (const float*)d_in[24];

  float* out    = (float*)d_out;
  float* Xpred  = out;                          // [B, T-1, N, D]
  float* out_ll = out + B_ * (T_ - 1) * N_ * D_;

  char* ws = (char*)d_ws;
  float*    Ap   = (float*)ws;    ws += (size_t)B_ * N_ * N_ * 4;   // 512 KB
  float*    hbf  = (float*)ws;    ws += (size_t)B_ * N_ * H_ * 4;   // 256 KB
  float*    agg  = (float*)ws;    ws += (size_t)B_ * N_ * H_ * 4;   // 256 KB
  _Float16* U    = (_Float16*)ws; ws += (size_t)B_ * N_ * MH_ * 2;  // 256 KB
  _Float16* V    = (_Float16*)ws; ws += (size_t)B_ * N_ * MH_ * 2;  // 256 KB
  _Float16* w1t  = (_Float16*)ws; ws += (size_t)2 * H_ * MH_ * 2;   // 32 KB
  float*    sqac = (float*)ws;    ws += 16;

  k_init<<<dim3(256), dim3(256), 0, stream>>>(hbf, sqac);
  k_prep<<<dim3(64), dim3(256), 0, stream>>>(Wm1, w1t);
  k_proj<<<dim3(N_, B_), dim3(N_), 0, stream>>>(A, Ap);

  for (int t = 0; t < T_ - 1; ++t) {
    k_uv<<<dim3(128), dim3(256), 0, stream>>>(hbf, w1t, U, V);
    k_msg<<<dim3(N_, B_), dim3(256), 0, stream>>>(U, V, Ap, bm1, Wm2, bm2, agg);
    k_gru<<<dim3(N_, B_), dim3(H_), 0, stream>>>(X, hbf, agg,
        Wir, bir, Wii, bii, Win, bin_,
        Whr, Whi, Whh, Wmr, Wmi, Wmn,
        Wo1, bo1, Wo2, bo2, Wo3, bo3,
        Xpred, sqac, t);
  }
  k_final<<<dim3(1), dim3(1), 0, stream>>>(log_sigma, sqac, out_ll);
}
